// Model_7928509628755
// MI455X (gfx1250) — matrix operand order repaired and checked by reading, not compiled, not run on hardware
//
#include <hip/hip_runtime.h>
#include <hip/hip_bf16.h>

// ---------------------------------------------------------------------------
// Types for WMMA fragments (wave32, gfx1250)
// ---------------------------------------------------------------------------
typedef __attribute__((ext_vector_type(16))) _Float16 v16h;
typedef __attribute__((ext_vector_type(8)))  _Float16 v8h;
typedef __attribute__((ext_vector_type(8)))  float    v8f;
typedef __attribute__((ext_vector_type(4)))  float    v4f;

#define N_NODES 6144
#define D_IN    256
#define HID     64
#define OUTD    32
#define ATT_H   128

// ---------------------------------------------------------------------------
// Convert + transpose: dst[c][r] = (f16) src[r][c].  dst is [cols x rows].
// Used to build the f16 B panels kept hot in L2.
// ---------------------------------------------------------------------------
__global__ void k_cvt_transpose_f16(const float* __restrict__ src,
                                    _Float16* __restrict__ dst,
                                    int rows, int cols) {
  int idx = blockIdx.x * blockDim.x + threadIdx.x;
  if (idx >= rows * cols) return;
  int r = idx / cols, c = idx - r * cols;
  dst[(size_t)c * rows + r] = (_Float16)src[(size_t)r * cols + c];
}

// ---------------------------------------------------------------------------
// One K=32 step of the 16x64 strip: A fragment (fp32->f16, NT stream) and
// 4 B fragments (f16 panel, cache-resident), 4 WMMAs.
// ---------------------------------------------------------------------------
__device__ __forceinline__ void wmma_kstep(
    const float* __restrict__ arow, int k, int half, int l16, int K,
    const _Float16* __restrict__ Bt, v8f acc[4])
{
  // ---- A fragment (ISA 16-bit A 16x32 layout) ----
  // lanes 0-15 : elems 0-7 = K k+0..7,  elems 8-15 = K k+16..23
  // lanes16-31 : elems 0-7 = K k+8..15, elems 8-15 = K k+24..31
  const float* pa = arow + k + half * 8;
  v4f a0 = __builtin_nontemporal_load((const v4f*)(pa));       // th:NT — 1 GB
  v4f a1 = __builtin_nontemporal_load((const v4f*)(pa + 4));   // stream must
  v4f a2 = __builtin_nontemporal_load((const v4f*)(pa + 16));  // not churn L2
  v4f a3 = __builtin_nontemporal_load((const v4f*)(pa + 20));
  v16h af;
#pragma unroll
  for (int i = 0; i < 4; ++i) {
    af[i]      = (_Float16)a0[i];
    af[4 + i]  = (_Float16)a1[i];
    af[8 + i]  = (_Float16)a2[i];
    af[12 + i] = (_Float16)a3[i];
  }

  // ---- B fragments: same K order as A (k+8h..+7 | k+16+8h..+7) ----
#pragma unroll
  for (int t = 0; t < 4; ++t) {
    const _Float16* pb = Bt + (size_t)(t * 16 + l16) * K + k + half * 8;
    v8h b0 = *(const v8h*)(pb);
    v8h b1 = *(const v8h*)(pb + 16);
    v16h bf;
#pragma unroll
    for (int i = 0; i < 8; ++i) { bf[i] = b0[i]; bf[8 + i] = b1[i]; }

    acc[t] = __builtin_amdgcn_wmma_f32_16x16x32_f16(
                 /*neg_a=*/false, af, /*neg_b=*/false, bf,
                 /*c_mod=*/(short)0, acc[t],
                 /*reuse_a=*/false, /*reuse_b=*/false);
  }
}

// ---------------------------------------------------------------------------
// C[M x 64] (+batch) = A_f32[M x K] @ Bt_f16[64 x K]^T   (f32 accumulate)
// Optionally C = tanh(C + bias).   K % 64 == 0, M % 128 == 0.
// 256 threads = 8 waves; wave owns a 16x64 strip (4 accumulators of 16x16).
// ---------------------------------------------------------------------------
template <int APPLY_BIAS_TANH>
__global__ __launch_bounds__(256) void wmma_gemm_n64(
    const float* __restrict__ Abase, size_t aBatchStride,
    const _Float16* __restrict__ Bt,     // [64 x K], row-contiguous per column
    const float* __restrict__ bias,      // [64] or nullptr
    float* __restrict__ Cbase, size_t cBatchStride,
    int M, int K)
{
  const int batch = blockIdx.y;
  const float* Ab = Abase + aBatchStride * (size_t)batch;
  float* C       = Cbase + cBatchStride * (size_t)batch;

  const int wave = threadIdx.x >> 5;       // 0..7
  const int lane = threadIdx.x & 31;
  const int half = lane >> 4;              // 0 | 1
  const int l16  = lane & 15;
  const int row0 = blockIdx.x * 128 + wave * 16;

  const float* arow = Ab + (size_t)(row0 + l16) * K;

  v8f acc[4] = {};                         // 4 x (16x16 f32) accumulators

  // K in chunks of 64: 8 WMMAs + 8 A-loads in flight per iteration for ILP.
  for (int k = 0; k < K; k += 64) {
    // Speculative prefetch two chunks ahead (dropped silently if OOB).
    __builtin_prefetch(arow + k + 128, 0, 1);
    wmma_kstep(arow, k,      half, l16, K, Bt, acc);
    wmma_kstep(arow, k + 32, half, l16, K, Bt, acc);
  }

  // ---- store: C/D layout — VGPR v, lane l -> row v + 8*(l>=16), col l&15 ----
#pragma unroll
  for (int t = 0; t < 4; ++t) {
    int col = t * 16 + l16;
#pragma unroll
    for (int v = 0; v < 8; ++v) {
      int row = row0 + v + half * 8;
      float val = acc[t][v];
      if (APPLY_BIAS_TANH) val = tanhf(val + bias[col]);
      C[(size_t)row * 64 + col] = val;
    }
  }
}

// ---------------------------------------------------------------------------
// Generic relation-aggregation stage (Dif_Classes_AGG / Weight_Schema /
// Classes_Relation_AGG):  per node n,
//   z[r] = tanh(Zin[r,n,:] @ W_r + b_r)            (HID=64)
//   s[r] = sum_h q[h] * tanh(z[r] @ aW + ab)[h]    (ATT_H=128)
//   out[n] = sum_r softmax(s)[r] * z[r]
// wStride/bStride = per-relation strides (0 => shared weight).
// One block of 64 threads per node.
// ---------------------------------------------------------------------------
__global__ __launch_bounds__(64) void k_relagg(
    const float* __restrict__ Zin, int Krel,
    const float* __restrict__ W, size_t wStride,
    const float* __restrict__ b, size_t bStride,
    const float* __restrict__ aW, const float* __restrict__ ab,
    const float* __restrict__ aq,
    float* __restrict__ out, int N)
{
  __shared__ float z[4][HID];
  __shared__ float red[HID];
  int n = blockIdx.x;
  int t = threadIdx.x;

  for (int r = 0; r < Krel; ++r) {
    const float* mrow = Zin + ((size_t)r * N + n) * HID;
    const float* Wr   = W + wStride * r;
    float acc = b[bStride * r + t];
    for (int d = 0; d < HID; ++d) acc += mrow[d] * Wr[d * HID + t];
    z[r][t] = tanhf(acc);
  }
  __syncthreads();

  float sloc[4];
  for (int r = 0; r < Krel; ++r) {
    float part = 0.f;
#pragma unroll
    for (int hh = 0; hh < 2; ++hh) {
      int h = t + hh * HID;
      float acc = ab[h];
      for (int d = 0; d < HID; ++d) acc += z[r][d] * aW[d * ATT_H + h];
      part += aq[h] * tanhf(acc);
    }
    red[t] = part;
    __syncthreads();
    for (int off = 32; off > 0; off >>= 1) {
      if (t < off) red[t] += red[t + off];
      __syncthreads();
    }
    sloc[r] = red[0];
    __syncthreads();
  }

  float mx = sloc[0];
  for (int r = 1; r < Krel; ++r) mx = fmaxf(mx, sloc[r]);
  float e[4], den = 0.f;
  for (int r = 0; r < Krel; ++r) { e[r] = __expf(sloc[r] - mx); den += e[r]; }
  float o = 0.f;
  for (int r = 0; r < Krel; ++r) o += (e[r] / den) * z[r][t];
  out[(size_t)n * HID + t] = o;
}

// ---------------------------------------------------------------------------
// Final: attention over the 2-stack {relation_agg, item_item} -> pooled,
// then logits = pooled @ W_out + b_out.  One 64-thread block per node.
// ---------------------------------------------------------------------------
__global__ __launch_bounds__(64) void k_final(
    const float* __restrict__ rel_agg, const float* __restrict__ item_item,
    const float* __restrict__ aW, const float* __restrict__ ab,
    const float* __restrict__ aq,
    const float* __restrict__ W_out, const float* __restrict__ b_out,
    float* __restrict__ logits, float* __restrict__ pooled_out, int N)
{
  __shared__ float z[2][HID];
  __shared__ float red[HID];
  __shared__ float pooled[HID];
  int n = blockIdx.x;
  int t = threadIdx.x;

  z[0][t] = rel_agg[(size_t)n * HID + t];
  z[1][t] = item_item[(size_t)n * HID + t];
  __syncthreads();

  float sloc[2];
  for (int r = 0; r < 2; ++r) {
    float part = 0.f;
#pragma unroll
    for (int hh = 0; hh < 2; ++hh) {
      int h = t + hh * HID;
      float acc = ab[h];
      for (int d = 0; d < HID; ++d) acc += z[r][d] * aW[d * ATT_H + h];
      part += aq[h] * tanhf(acc);
    }
    red[t] = part;
    __syncthreads();
    for (int off = 32; off > 0; off >>= 1) {
      if (t < off) red[t] += red[t + off];
      __syncthreads();
    }
    sloc[r] = red[0];
    __syncthreads();
  }

  float mx = fmaxf(sloc[0], sloc[1]);
  float e0 = __expf(sloc[0] - mx), e1 = __expf(sloc[1] - mx);
  float inv = 1.f / (e0 + e1);
  float p = (e0 * inv) * z[0][t] + (e1 * inv) * z[1][t];
  pooled[t] = p;
  pooled_out[(size_t)n * HID + t] = p;
  __syncthreads();

  if (t < OUTD) {
    float acc = b_out[t];
    for (int d = 0; d < HID; ++d) acc += pooled[d] * W_out[d * OUTD + t];
    logits[(size_t)n * OUTD + t] = acc;
  }
}

// ---------------------------------------------------------------------------
extern "C" void kernel_launch(void* const* d_in, const int* in_sizes, int n_in,
                              void* d_out, int out_size, void* d_ws, size_t ws_size,
                              hipStream_t stream) {
  const int N = N_NODES;
  const float* feat_item = (const float*)d_in[0];
  const float* feat_user = (const float*)d_in[1];
  const float* adj_ii    = (const float*)d_in[2];   // [2,N,N]
  const float* adj_mp    = (const float*)d_in[3];   // [3,N,N]
  const float* adj_ui    = (const float*)d_in[4];   // [2,N,N]
  const float* W_map0    = (const float*)d_in[5];
  const float* b_map0    = (const float*)d_in[6];
  const float* W_map1    = (const float*)d_in[7];
  const float* b_map1    = (const float*)d_in[8];
  const float* W_rel     = (const float*)d_in[9];
  const float* b_rel     = (const float*)d_in[10];
  const float* rel_aW    = (const float*)d_in[11];
  const float* rel_ab    = (const float*)d_in[12];
  const float* rel_aq    = (const float*)d_in[13];
  const float* W_sch     = (const float*)d_in[14];
  const float* b_sch     = (const float*)d_in[15];
  const float* sch_aW    = (const float*)d_in[16];
  const float* sch_ab    = (const float*)d_in[17];
  const float* sch_aq    = (const float*)d_in[18];
  const float* W_ii      = (const float*)d_in[19];
  const float* b_ii      = (const float*)d_in[20];
  const float* ii_aW     = (const float*)d_in[21];
  const float* ii_ab     = (const float*)d_in[22];
  const float* ii_aq     = (const float*)d_in[23];
  const float* fin_aW    = (const float*)d_in[24];
  const float* fin_ab    = (const float*)d_in[25];
  const float* fin_aq    = (const float*)d_in[26];
  const float* W_out     = (const float*)d_in[27];
  const float* b_out     = (const float*)d_in[28];

  // --- carve workspace (256 B aligned) ---
  char* ws = (char*)d_ws;
  auto carve = [&](size_t bytes) {
    char* p = ws; ws += (bytes + 255) & ~(size_t)255; return p;
  };
  float*    h0    = (float*)   carve((size_t)N * HID * 4);
  float*    h1    = (float*)   carve((size_t)N * HID * 4);
  _Float16* Wt0   = (_Float16*)carve((size_t)D_IN * HID * 2);
  _Float16* Wt1   = (_Float16*)carve((size_t)D_IN * HID * 2);
  _Float16* h0t   = (_Float16*)carve((size_t)HID * N * 2);
  _Float16* h1t   = (_Float16*)carve((size_t)HID * N * 2);
  _Float16* ict   = (_Float16*)carve((size_t)HID * N * 2);
  float*    m     = (float*)   carve((size_t)3 * N * HID * 4);
  float*    sbuf  = (float*)   carve((size_t)2 * N * HID * 4);
  float*    tbuf  = (float*)   carve((size_t)2 * N * HID * 4);
  float*    ragg  = (float*)   carve((size_t)N * HID * 4);
  float*    iclas = (float*)   carve((size_t)N * HID * 4);
  float*    iitem = (float*)   carve((size_t)N * HID * 4);

  const size_t NN = (size_t)N * N;
  const size_t NH = (size_t)N * HID;
  const dim3 blk(256);

  // 1) f16-transpose the mapping weights (B panels for the map GEMMs)
  int nW = D_IN * HID;
  k_cvt_transpose_f16<<<(nW + 255) / 256, 256, 0, stream>>>(W_map0, Wt0, D_IN, HID);
  k_cvt_transpose_f16<<<(nW + 255) / 256, 256, 0, stream>>>(W_map1, Wt1, D_IN, HID);

  // 2) feature mapping: h = tanh(X @ Wmap + b)   [WMMA, K=256]
  wmma_gemm_n64<1><<<dim3(N / 128, 1), blk, 0, stream>>>(feat_item, 0, Wt0, b_map0, h0, 0, N, D_IN);
  wmma_gemm_n64<1><<<dim3(N / 128, 1), blk, 0, stream>>>(feat_user, 0, Wt1, b_map1, h1, 0, N, D_IN);

  // 3) f16-transpose h panels (resident in L2 for the big GEMMs)
  k_cvt_transpose_f16<<<((int)NH + 255) / 256, 256, 0, stream>>>(h0, h0t, N, HID);
  k_cvt_transpose_f16<<<((int)NH + 255) / 256, 256, 0, stream>>>(h1, h1t, N, HID);

  // 4) big adjacency GEMMs (bandwidth-critical; WMMA f16->f32, K=6144)
  wmma_gemm_n64<0><<<dim3(N / 128, 3), blk, 0, stream>>>(adj_mp, NN, h0t, nullptr, m,    NH, N, N);
  wmma_gemm_n64<0><<<dim3(N / 128, 2), blk, 0, stream>>>(adj_ui, NN, h1t, nullptr, sbuf, NH, N, N);

  // 5) relation / schema aggregation stages
  k_relagg<<<N, 64, 0, stream>>>(m,    3, W_rel, (size_t)HID * HID, b_rel, HID,
                                 rel_aW, rel_ab, rel_aq, ragg, N);
  k_relagg<<<N, 64, 0, stream>>>(sbuf, 2, W_sch, 0, b_sch, 0,
                                 sch_aW, sch_ab, sch_aq, iclas, N);

  // 6) item-item GEMMs on item_classes
  k_cvt_transpose_f16<<<((int)NH + 255) / 256, 256, 0, stream>>>(iclas, ict, N, HID);
  wmma_gemm_n64<0><<<dim3(N / 128, 2), blk, 0, stream>>>(adj_ii, NN, ict, nullptr, tbuf, NH, N, N);
  k_relagg<<<N, 64, 0, stream>>>(tbuf, 2, W_ii, (size_t)HID * HID, b_ii, HID,
                                 ii_aW, ii_ab, ii_aq, iitem, N);

  // 7) final attention + projection -> d_out = [logits N*32 | pooled N*64]
  float* logits = (float*)d_out;
  float* pooled = (float*)d_out + (size_t)N * OUTD;
  k_final<<<N, 64, 0, stream>>>(ragg, iitem, fin_aW, fin_ab, fin_aq,
                                W_out, b_out, logits, pooled, N);
}
